// MambaTab_47614007443811
// MI455X (gfx1250) — matrix operand order repaired and checked by reading, not compiled, not run on hardware
//
#include <hip/hip_runtime.h>
#include <math.h>

// ---- problem constants (match reference) ----
#define BB 8
#define LL 1024
#define FF 128
#define DD 256          // d_model
#define DIN 512         // d_inner
#define NSt 32          // d_state
#define KC 4            // d_conv
#define RR 16           // dt_rank
#define NCp 2           // n_class
#define BL (BB*LL)      // 8192 token rows
#define DBCP 128        // padded width of dbc (dt:0..15, B:16..47, C:48..79, pad:80..127)

typedef __attribute__((ext_vector_type(16))) __bf16 v16bf;
typedef __attribute__((ext_vector_type(8)))  __bf16 v8bf;
typedef __attribute__((ext_vector_type(8)))  float  v8f;

__device__ __forceinline__ float sigf(float x) { return 1.f / (1.f + __expf(-x)); }

// ---------------------------------------------------------------------------
// f32 -> bf16 staging convert (flat)
// ---------------------------------------------------------------------------
__global__ void cvt_bf16(const float* __restrict__ in, __bf16* __restrict__ out, int n) {
    int i = blockIdx.x * blockDim.x + threadIdx.x;
    if (i < n) out[i] = (__bf16)in[i];
}

// ---------------------------------------------------------------------------
// f32 -> bf16 with 2-D zero padding:
// dst[r, c] (dst_rows x dst_cols) = (r<src_rows && c<src_cols) ? src[r*src_ld+c] : 0
// ---------------------------------------------------------------------------
__global__ void cvt_pad2d(const float* __restrict__ src, __bf16* __restrict__ dst,
                          int dst_rows, int dst_cols, int src_rows, int src_cols, int src_ld)
{
    int i = blockIdx.x * blockDim.x + threadIdx.x;
    if (i >= dst_rows * dst_cols) return;
    int r = i / dst_cols, c = i % dst_cols;
    dst[i] = (r < src_rows && c < src_cols) ? (__bf16)src[(size_t)r * src_ld + c] : (__bf16)0.f;
}

// ---------------------------------------------------------------------------
// WMMA bf16 GEMM: C[M,N] = A[M,K](row-major,lda) * W[N,K]^T(row-major,ldw)
// K: compile-time, multiple of 32 (pad inputs otherwise).
// Wave computes a 16(M) x 64(N) strip (4 WMMA tiles sharing the A fragment);
// block = 4 waves stacked along M; grid = (N/64, M/64).
// Fragment layouts used here (wave32; contiguous k order in A and B, not the ISA table order):
//   A lane: one contiguous 16-elem run  K=[k0+16h, +16), row m per lane
//   B lane: one contiguous 16-elem run  K=[k0+16h, +16), column n per lane
//   D lane: VGPR r -> row m0 + r + 8h, col n0 + (lane&15)
// ACT: 0 = +bias ; 1 = +bias then softplus
// ---------------------------------------------------------------------------
template<int K, int ACT>
__global__ void wmma_gemm(const __bf16* __restrict__ A, const __bf16* __restrict__ W,
                          const float* __restrict__ bias, float* __restrict__ C,
                          int lda, int ldw, int ldc)
{
    const int lane = threadIdx.x;      // 0..31
    const int half = lane >> 4;        // 0 or 1
    const int ln16 = lane & 15;
    const int m0 = (blockIdx.y * 4 + threadIdx.y) * 16;
    const int n0 = blockIdx.x * 64;

    const __bf16* __restrict__ Arow = A + (size_t)(m0 + ln16) * lda;
    const __bf16* __restrict__ Wc0  = W + (size_t)(n0 +  0 + ln16) * ldw;
    const __bf16* __restrict__ Wc1  = W + (size_t)(n0 + 16 + ln16) * ldw;
    const __bf16* __restrict__ Wc2  = W + (size_t)(n0 + 32 + ln16) * ldw;
    const __bf16* __restrict__ Wc3  = W + (size_t)(n0 + 48 + ln16) * ldw;

    v8f acc0 = {}, acc1 = {}, acc2 = {}, acc3 = {};

#pragma unroll
    for (int k0 = 0; k0 < K; k0 += 32) {
        // A fragment: one 32-byte contiguous load
        v16bf a = *(const v16bf*)(Arow + k0 + 16 * half);
        // B fragments: one 32-byte contiguous load per N-tile
        v16bf b0 = *(const v16bf*)(Wc0 + k0 + 16 * half);
        v16bf b1 = *(const v16bf*)(Wc1 + k0 + 16 * half);
        v16bf b2 = *(const v16bf*)(Wc2 + k0 + 16 * half);
        v16bf b3 = *(const v16bf*)(Wc3 + k0 + 16 * half);
        acc0 = __builtin_amdgcn_wmma_f32_16x16x32_bf16(false, a, false, b0, (short)0, acc0, false, false);
        acc1 = __builtin_amdgcn_wmma_f32_16x16x32_bf16(false, a, false, b1, (short)0, acc1, false, false);
        acc2 = __builtin_amdgcn_wmma_f32_16x16x32_bf16(false, a, false, b2, (short)0, acc2, false, false);
        acc3 = __builtin_amdgcn_wmma_f32_16x16x32_bf16(false, a, false, b3, (short)0, acc3, false, false);
    }

    const v8f accs[4] = { acc0, acc1, acc2, acc3 };
#pragma unroll
    for (int t = 0; t < 4; ++t) {
        const int col = n0 + t * 16 + ln16;
        const float bv = bias ? bias[col] : 0.f;
#pragma unroll
        for (int r = 0; r < 8; ++r) {
            const int row = m0 + r + (half ? 8 : 0);
            float v = accs[t][r] + bv;
            if (ACT == 1) v = (v > 20.f) ? v : log1pf(__expf(v));  // softplus
            C[(size_t)row * ldc + col] = v;
        }
    }
}

// ---------------------------------------------------------------------------
// LayerNorm + ReLU over D=256, emit bf16. One 256-thread block per row.
// ---------------------------------------------------------------------------
__global__ void ln_relu_bf16(const float* __restrict__ h, const float* __restrict__ g,
                             const float* __restrict__ beta, __bf16* __restrict__ out)
{
    const int row = blockIdx.x;
    const int tid = threadIdx.x;           // 0..255
    float v = h[(size_t)row * DD + tid];
    float s = v, s2 = v * v;
#pragma unroll
    for (int off = 16; off; off >>= 1) { s += __shfl_xor(s, off); s2 += __shfl_xor(s2, off); }
    __shared__ float rs[8], rq[8];
    const int wid = tid >> 5, lane = tid & 31;
    if (!lane) { rs[wid] = s; rq[wid] = s2; }
    __syncthreads();
    if (tid == 0) {
        float a = 0.f, b = 0.f;
        for (int i = 0; i < 8; ++i) { a += rs[i]; b += rq[i]; }
        rs[0] = a; rq[0] = b;
    }
    __syncthreads();
    const float mu   = rs[0] * (1.f / DD);
    const float var  = rq[0] * (1.f / DD) - mu * mu;
    const float rstd = rsqrtf(var + 1e-5f);
    float y = (v - mu) * rstd * g[tid] + beta[tid];
    y = fmaxf(y, 0.f);
    out[(size_t)row * DD + tid] = (__bf16)y;
}

// ---------------------------------------------------------------------------
// Depthwise causal conv1d (K=4) + SiLU on xi = xz[:, :DIN]; emit f32 + bf16
// ---------------------------------------------------------------------------
__global__ void conv_silu(const float* __restrict__ xz, const float* __restrict__ cw,
                          const float* __restrict__ cb, float* __restrict__ xa,
                          __bf16* __restrict__ xab)
{
    int i = blockIdx.x * blockDim.x + threadIdx.x;   // over BL*DIN
    if (i >= BL * DIN) return;
    const int d  = i % DIN;
    const int bl = i / DIN;
    const int l  = bl % LL;
    float acc = cb[d];
#pragma unroll
    for (int j = 0; j < KC; ++j) {
        int ls = l - (KC - 1) + j;
        if (ls >= 0)
            acc += cw[d * KC + j] * xz[(size_t)(bl - (KC - 1) + j) * (2 * DIN) + d];
    }
    float s = acc * sigf(acc);
    xa[i]  = s;
    xab[i] = (__bf16)s;
}

// ---------------------------------------------------------------------------
// Selective scan: one wave32 per (b, d) channel; lane n owns state n (N=32).
// ---------------------------------------------------------------------------
__global__ void selective_scan(const float* __restrict__ delta, const float* __restrict__ xa,
                               const float* __restrict__ dbc, const float* __restrict__ A_log,
                               float* __restrict__ ysc)
{
    const int w = blockIdx.x * blockDim.y + threadIdx.y;   // wave id over B*DIN
    if (w >= BB * DIN) return;
    const int n = threadIdx.x;                              // state index 0..31
    const int b = w / DIN, d = w % DIN;

    const float Av = -__expf(A_log[d * NSt + n]);
    float h = 0.f;
    const float* __restrict__ dl  = delta + (size_t)b * LL * DIN + d;
    const float* __restrict__ xp  = xa    + (size_t)b * LL * DIN + d;
    const float* __restrict__ dbp = dbc   + (size_t)b * LL * DBCP;
    float* __restrict__ yp        = ysc   + (size_t)b * LL * DIN + d;

    for (int l = 0; l < LL; ++l) {
        const float dt = dl[(size_t)l * DIN];
        const float xi = xp[(size_t)l * DIN];
        const float Bv = dbp[l * DBCP + RR + n];
        const float Cv = dbp[l * DBCP + RR + NSt + n];
        h = __expf(dt * Av) * h + dt * Bv * xi;
        float p = h * Cv;
#pragma unroll
        for (int off = 16; off; off >>= 1) p += __shfl_xor(p, off);
        if (n == 0) yp[(size_t)l * DIN] = p;
    }
}

// ---------------------------------------------------------------------------
// Gate: y = (scan + xi*D) * silu(z)
// ---------------------------------------------------------------------------
__global__ void gate_kernel(const float* __restrict__ ysc, const float* __restrict__ xa,
                            const float* __restrict__ Dp, const float* __restrict__ xz,
                            float* __restrict__ y)
{
    int i = blockIdx.x * blockDim.x + threadIdx.x;
    if (i >= BL * DIN) return;
    const int d = i % DIN;
    const int row = i / DIN;
    const float z = xz[(size_t)row * (2 * DIN) + DIN + d];
    y[i] = (ysc[i] + xa[i] * Dp[d]) * (z * sigf(z));
}

// ---------------------------------------------------------------------------
// Fold classifier into out_proj: M2[c, dd] = sum_d W_out[c,d] * out_proj[d,dd]
// ---------------------------------------------------------------------------
__global__ void fold_w(const float* __restrict__ Wout, const float* __restrict__ Op,
                       float* __restrict__ M2)
{
    int i = blockIdx.x * blockDim.x + threadIdx.x;  // NCp*DIN = 1024
    if (i >= NCp * DIN) return;
    const int c = i / DIN, dd = i % DIN;
    float s = 0.f;
    for (int d = 0; d < DD; ++d) s += Wout[c * DD + d] * Op[d * DIN + dd];
    M2[i] = s;
}

// ---------------------------------------------------------------------------
// Final: out[row,c] = b_out[c] + sum_dd y[row,dd]*M2[c,dd]  (wave per (row,c))
// ---------------------------------------------------------------------------
__global__ void final_out(const float* __restrict__ y, const float* __restrict__ M2,
                          const float* __restrict__ bout, float* __restrict__ out)
{
    const int w = blockIdx.x * blockDim.y + threadIdx.y;  // over BL*NCp
    if (w >= BL * NCp) return;
    const int row = w / NCp, c = w % NCp;
    const int lane = threadIdx.x;
    float s = 0.f;
    for (int d = lane; d < DIN; d += 32)
        s += y[(size_t)row * DIN + d] * M2[c * DIN + d];
#pragma unroll
    for (int off = 16; off; off >>= 1) s += __shfl_xor(s, off);
    if (!lane) out[row * NCp + c] = s + bout[c];
}

// ---------------------------------------------------------------------------
extern "C" void kernel_launch(void* const* d_in, const int* in_sizes, int n_in,
                              void* d_out, int out_size, void* d_ws, size_t ws_size,
                              hipStream_t stream)
{
    (void)in_sizes; (void)n_in; (void)out_size; (void)ws_size;

    const float* x         = (const float*)d_in[0];
    const float* W1        = (const float*)d_in[1];
    const float* b1        = (const float*)d_in[2];
    const float* ln_g      = (const float*)d_in[3];
    const float* ln_b      = (const float*)d_in[4];
    const float* in_proj_w = (const float*)d_in[5];
    const float* conv_w    = (const float*)d_in[6];
    const float* conv_b    = (const float*)d_in[7];
    const float* x_proj_w  = (const float*)d_in[8];
    const float* dt_proj_w = (const float*)d_in[9];
    const float* dt_proj_b = (const float*)d_in[10];
    const float* A_log     = (const float*)d_in[11];
    const float* D_param   = (const float*)d_in[12];
    const float* out_proj  = (const float*)d_in[13];
    const float* W_out     = (const float*)d_in[14];
    const float* b_out     = (const float*)d_in[15];
    float* out             = (float*)d_out;

    char* ws = (char*)d_ws;
    size_t off = 0;
    auto alloc = [&](size_t bytes) -> char* {
        char* p = ws + off;
        off += (bytes + 255) & ~(size_t)255;
        return p;
    };

    __bf16* x_bf    = (__bf16*)alloc((size_t)BL * FF * 2);
    __bf16* W1_bf   = (__bf16*)alloc((size_t)DD * FF * 2);
    __bf16* inW_bf  = (__bf16*)alloc((size_t)2 * DIN * DD * 2);
    __bf16* xpW_bf  = (__bf16*)alloc((size_t)DBCP * DIN * 2);   // padded to 128 rows
    __bf16* dtW_bf  = (__bf16*)alloc((size_t)DIN * 32 * 2);     // K padded 16->32
    float*  h1      = (float*) alloc((size_t)BL * DD * 4);
    __bf16* u_bf    = (__bf16*)alloc((size_t)BL * DD * 2);
    float*  xz      = (float*) alloc((size_t)BL * 2 * DIN * 4);
    float*  xa      = (float*) alloc((size_t)BL * DIN * 4);
    __bf16* xa_bf   = (__bf16*)alloc((size_t)BL * DIN * 2);
    float*  dbc     = (float*) alloc((size_t)BL * DBCP * 4);    // padded width 128
    __bf16* dt_bf   = (__bf16*)alloc((size_t)BL * 32 * 2);      // dt cols, K padded
    float*  delta   = (float*) alloc((size_t)BL * DIN * 4);
    float*  ysc     = (float*) alloc((size_t)BL * DIN * 4);
    float*  yfin    = (float*) alloc((size_t)BL * DIN * 4);
    float*  M2      = (float*) alloc((size_t)NCp * DIN * 4);

    auto cvt = [&](const float* src, __bf16* dst, int n) {
        cvt_bf16<<<(n + 255) / 256, 256, 0, stream>>>(src, dst, n);
    };
    auto pad2d = [&](const float* src, __bf16* dst, int dr, int dc, int sr, int sc, int sld) {
        cvt_pad2d<<<(dr * dc + 255) / 256, 256, 0, stream>>>(src, dst, dr, dc, sr, sc, sld);
    };

    // stage bf16 copies (weights once; cheap relative to pipeline)
    cvt(x, x_bf, BL * FF);
    cvt(W1, W1_bf, DD * FF);
    cvt(in_proj_w, inW_bf, 2 * DIN * DD);
    pad2d(x_proj_w, xpW_bf, DBCP, DIN, RR + 2 * NSt, DIN, DIN);  // pad 80 -> 128 rows
    pad2d(dt_proj_w, dtW_bf, DIN, 32, DIN, RR, RR);              // pad K 16 -> 32

    const dim3 gblk(32, 4);   // 4 waves / block; block covers 64M x 64N

    // 1) h1 = x @ W1^T + b1
    wmma_gemm<FF, 0><<<dim3(DD / 64, BL / 64), gblk, 0, stream>>>(
        x_bf, W1_bf, b1, h1, FF, FF, DD);

    // 2) LayerNorm + ReLU -> u (bf16)
    ln_relu_bf16<<<BL, DD, 0, stream>>>(h1, ln_g, ln_b, u_bf);

    // 3) xz = u @ in_proj^T
    wmma_gemm<DD, 0><<<dim3((2 * DIN) / 64, BL / 64), gblk, 0, stream>>>(
        u_bf, inW_bf, nullptr, xz, DD, DD, 2 * DIN);

    // 4) depthwise causal conv + SiLU
    conv_silu<<<(BL * DIN + 255) / 256, 256, 0, stream>>>(xz, conv_w, conv_b, xa, xa_bf);

    // 5) dbc = xi @ x_proj^T   (N padded to 128)
    wmma_gemm<DIN, 0><<<dim3(DBCP / 64, BL / 64), gblk, 0, stream>>>(
        xa_bf, xpW_bf, nullptr, dbc, DIN, DIN, DBCP);

    // 6) dt columns -> bf16 with K padded to 32, then
    //    delta = softplus(dt @ dt_proj^T + dt_proj_b)
    pad2d(dbc, dt_bf, BL, 32, BL, RR, DBCP);
    wmma_gemm<32, 1><<<dim3(DIN / 64, BL / 64), gblk, 0, stream>>>(
        dt_bf, dtW_bf, dt_proj_b, delta, 32, 32, DIN);

    // 7) selective scan: 4096 waves, 8 waves/block
    selective_scan<<<(BB * DIN) / 8, dim3(32, 8), 0, stream>>>(delta, xa, dbc, A_log, ysc);

    // 8) gating
    gate_kernel<<<(BL * DIN + 255) / 256, 256, 0, stream>>>(ysc, xa, D_param, xz, yfin);

    // 9) fold W_out @ out_proj -> M2 (2 x 512), then fused output GEMV
    fold_w<<<(NCp * DIN + 255) / 256, 256, 0, stream>>>(W_out, out_proj, M2);
    final_out<<<(BL * NCp) / 8, dim3(32, 8), 0, stream>>>(yfin, M2, b_out, out);
}
